// HardMoEProjection_7284264534308
// MI455X (gfx1250) — matrix operand order repaired and checked by reading, not compiled, not run on hardware
//
#include <hip/hip_runtime.h>

// ---------------------------------------------------------------------------
// HardMoE top-1 projection for MI455X (gfx1250, wave32, WMMA bf16 path)
//   B=8192, D_IN=1024, D_HID=2048, D_OUT=2048, E=8
// ---------------------------------------------------------------------------

#define B_ROWS   8192
#define D_IN     1024
#define D_HID    2048
#define D_OUT    2048
#define N_EXP    8
#define PERM_N   8320            // 8192 + 8*16 padding, multiple of 16
#define N_TILES  (PERM_N / 16)   // 520

typedef __attribute__((ext_vector_type(16))) __bf16 v16bf;
typedef __attribute__((ext_vector_type(8)))  __bf16 v8bf;
typedef __attribute__((ext_vector_type(8)))  float  v8f;

union AFrag { v16bf v; v8bf h[2]; };

__device__ __forceinline__ v8f wmma_bf16(v16bf a, v16bf b, v8f c) {
  // D = A(16x32 bf16) * B(32x16 bf16) + C(16x16 f32)
  return __builtin_amdgcn_wmma_f32_16x16x32_bf16(
      /*neg_a=*/false, a, /*neg_b=*/false, b,
      /*c_mod=*/(short)0, c, /*reuse_a=*/false, /*reuse_b=*/false);
}

// ---------------------------------------------------------------------------
// Kernel 1: fp32 [K,N] row-major  ->  bf16 [N,K] row-major (K-contiguous)
// 32x32 tiles through LDS; coalesced on both sides.
// ---------------------------------------------------------------------------
__global__ __launch_bounds__(256)
void k_transpose_bf16(const float* __restrict__ in, __bf16* __restrict__ out,
                      int K, int N) {
  __shared__ float s[32][33];
  const int n0 = blockIdx.x * 32;
  const int k0 = blockIdx.y * 32;
  const int tx = threadIdx.x & 31;   // inner (coalesced) index
  const int ty = threadIdx.x >> 5;   // 8 rows per pass
#pragma unroll
  for (int p = 0; p < 4; ++p) {
    const int k = k0 + ty + p * 8;
    s[ty + p * 8][tx] = in[(size_t)k * N + n0 + tx];
  }
  __syncthreads();
#pragma unroll
  for (int p = 0; p < 4; ++p) {
    const int n = n0 + ty + p * 8;
    out[(size_t)n * K + k0 + tx] = (__bf16)s[tx][ty + p * 8];
  }
}

// ---------------------------------------------------------------------------
// Kernel 2: init counters / cursors / permutation sentinels
// ---------------------------------------------------------------------------
__global__ void k_init(int* __restrict__ cnt16, int* __restrict__ perm) {
  const int i = blockIdx.x * blockDim.x + threadIdx.x;
  if (i < 16) cnt16[i] = 0;          // [0..7]=counts, [8..15]=cursors
  if (i < PERM_N) perm[i] = -1;
}

// ---------------------------------------------------------------------------
// Kernel 3: router.  One WG (256 thr = 8 waves) per 16-row tile.
//   h = relu(x @ Wr1 + br1); scores = h @ Wr2 + br2; idx = argmax(scores)
// Wave w covers hidden columns [w*256, w*256+256) as 16 WMMA N-subtiles.
// ---------------------------------------------------------------------------
__global__ __launch_bounds__(256)
void k_router(const float* __restrict__ x,
              const __bf16* __restrict__ Wr1T,   // [D_HID][D_IN] bf16
              const float* __restrict__ br1,
              const float* __restrict__ Wr2,     // [D_HID][N_EXP] fp32
              const float* __restrict__ br2,
              int* __restrict__ idx_out) {
  __shared__ __bf16 sA[16 * D_IN];               // 32 KB  A tile (rows x K)
  __shared__ float  sScores[16 * N_EXP];

  const int b0 = blockIdx.x * 16;

  // Stage A tile (fp32 -> bf16), coalesced along K.
  for (int i = threadIdx.x; i < 16 * D_IN; i += 256) {
    const int m = i >> 10, k = i & (D_IN - 1);
    sA[i] = (__bf16)x[(size_t)(b0 + m) * D_IN + k];
  }
  if (threadIdx.x < 16 * N_EXP) sScores[threadIdx.x] = 0.0f;
  __syncthreads();

  const int wave = threadIdx.x >> 5;
  const int lane = threadIdx.x & 31;
  const int half = lane >> 4;        // 0: lanes 0-15, 1: lanes 16-31
  const int mrow = lane & 15;        // A row / C column within tile

  float ps[8][N_EXP] = {};           // per-lane partial scores (reg)

  for (int nt = 0; nt < 16; ++nt) {
    const int col = wave * 256 + nt * 16 + mrow;          // hidden column
    const __bf16* bptr = Wr1T + (size_t)col * D_IN;
    v8f c = {0.f, 0.f, 0.f, 0.f, 0.f, 0.f, 0.f, 0.f};
    for (int kk = 0; kk < D_IN; kk += 32) {
      if (kk + 128 < D_IN)
        __builtin_prefetch((const void*)(bptr + kk + 128), 0, 0);
      AFrag a;
      a.h[0] = *(const v8bf*)(&sA[mrow * D_IN + kk + half * 8]);
      a.h[1] = *(const v8bf*)(&sA[mrow * D_IN + kk + 16 + half * 8]);
      AFrag bf;
      bf.h[0] = *(const v8bf*)(bptr + kk + half * 8);
      bf.h[1] = *(const v8bf*)(bptr + kk + 16 + half * 8);
      const v16bf b = bf.v;
      c = wmma_bf16(a.v, b, c);
    }
    // bias + relu + fold into score partials (scores = relu(h) @ Wr2)
    const float bias = br1[col];
    float w2[N_EXP];
#pragma unroll
    for (int e = 0; e < N_EXP; ++e) w2[e] = Wr2[(size_t)col * N_EXP + e];
#pragma unroll
    for (int r = 0; r < 8; ++r) {
      float h = c[r] + bias;
      h = h > 0.0f ? h : 0.0f;
#pragma unroll
      for (int e = 0; e < N_EXP; ++e) ps[r][e] += h * w2[e];
    }
  }

  // Reduce partials across the 16 lanes sharing the same M, then LDS atomic.
#pragma unroll
  for (int r = 0; r < 8; ++r) {
#pragma unroll
    for (int e = 0; e < N_EXP; ++e) {
      float v = ps[r][e];
      v += __shfl_xor(v, 1, 32);
      v += __shfl_xor(v, 2, 32);
      v += __shfl_xor(v, 4, 32);
      v += __shfl_xor(v, 8, 32);
      if ((lane & 15) == 0)
        atomicAdd(&sScores[(r + 8 * half) * N_EXP + e], v);
    }
  }
  __syncthreads();

  if (threadIdx.x < 16) {
    const int m = threadIdx.x;
    float best = -3.402823466e+38f;
    int bi = 0;
#pragma unroll
    for (int e = 0; e < N_EXP; ++e) {
      const float s = sScores[m * N_EXP + e] + br2[e];
      if (s > best) { best = s; bi = e; }   // first-max tie-break (argmax)
    }
    idx_out[b0 + m] = bi;
  }
}

// ---------------------------------------------------------------------------
// Kernels 4-6: counting sort of rows by expert, segments padded to x16.
// ---------------------------------------------------------------------------
__global__ void k_count(const int* __restrict__ idx, int* __restrict__ cnt) {
  const int i = blockIdx.x * blockDim.x + threadIdx.x;
  if (i < B_ROWS) atomicAdd(&cnt[idx[i]], 1);
}

__global__ void k_offsets(int* __restrict__ cnt16) {
  int off = 0;
#pragma unroll
  for (int e = 0; e < N_EXP; ++e) {
    cnt16[8 + e] = off;                       // cursor
    off += (cnt16[e] + 15) & ~15;             // padded segment
  }
}

__global__ void k_scatter(const int* __restrict__ idx, int* __restrict__ cnt16,
                          int* __restrict__ perm) {
  const int i = blockIdx.x * blockDim.x + threadIdx.x;
  if (i < B_ROWS) {
    const int e = idx[i];
    const int p = atomicAdd(&cnt16[8 + e], 1);
    perm[p] = i;
  }
}

// ---------------------------------------------------------------------------
// Kernel 7: grouped expert GEMM.  One WG per padded 16-row tile (all rows
// share one expert).  out[row] = x[row] @ We[:, e*D_OUT : ...] + be[e]
// ---------------------------------------------------------------------------
__global__ __launch_bounds__(256)
void k_expert(const float* __restrict__ x,
              const __bf16* __restrict__ WeT,    // [E*D_OUT][D_IN] bf16
              const float* __restrict__ be,      // [E*D_OUT]
              const int* __restrict__ idx,
              const int* __restrict__ perm,
              float* __restrict__ out) {
  __shared__ __bf16 sA[16 * D_IN];
  __shared__ int sRows[16];
  __shared__ int sExpert;

  const int t = blockIdx.x;
  if (threadIdx.x == 0) {
    int e = -1;
#pragma unroll
    for (int j = 0; j < 16; ++j) {
      const int r = perm[t * 16 + j];
      sRows[j] = r;
      if (e < 0 && r >= 0) e = idx[r];
    }
    sExpert = e;
  }
  __syncthreads();
  if (sExpert < 0) return;                     // fully-padded tile (uniform)

  for (int i = threadIdx.x; i < 16 * D_IN; i += 256) {
    const int m = i >> 10, k = i & (D_IN - 1);
    const int row = sRows[m];
    sA[i] = (row >= 0) ? (__bf16)x[(size_t)row * D_IN + k] : (__bf16)0.0f;
  }
  __syncthreads();

  const int e    = sExpert;
  const int wave = threadIdx.x >> 5;
  const int lane = threadIdx.x & 31;
  const int half = lane >> 4;
  const int mrow = lane & 15;

  for (int nt = 0; nt < 16; ++nt) {
    const int col  = wave * 256 + nt * 16 + mrow;          // 0..2047
    const int colg = e * D_OUT + col;                      // global weight col
    const __bf16* bptr = WeT + (size_t)colg * D_IN;
    v8f c = {0.f, 0.f, 0.f, 0.f, 0.f, 0.f, 0.f, 0.f};
    for (int kk = 0; kk < D_IN; kk += 32) {
      if (kk + 128 < D_IN)
        __builtin_prefetch((const void*)(bptr + kk + 128), 0, 0);
      AFrag a;
      a.h[0] = *(const v8bf*)(&sA[mrow * D_IN + kk + half * 8]);
      a.h[1] = *(const v8bf*)(&sA[mrow * D_IN + kk + 16 + half * 8]);
      AFrag bf;
      bf.h[0] = *(const v8bf*)(bptr + kk + half * 8);
      bf.h[1] = *(const v8bf*)(bptr + kk + 16 + half * 8);
      const v16bf b = bf.v;
      c = wmma_bf16(a.v, b, c);
    }
    const float bias = be[colg];
#pragma unroll
    for (int r = 0; r < 8; ++r) {
      const int m   = r + 8 * half;
      const int row = sRows[m];
      if (row >= 0) out[(size_t)row * D_OUT + col] = c[r] + bias;
    }
  }
}

// ---------------------------------------------------------------------------
// Host launcher
// ---------------------------------------------------------------------------
extern "C" void kernel_launch(void* const* d_in, const int* in_sizes, int n_in,
                              void* d_out, int out_size, void* d_ws, size_t ws_size,
                              hipStream_t stream) {
  const float* x   = (const float*)d_in[0];   // [B, D_IN]
  const float* We  = (const float*)d_in[1];   // [D_IN, E*D_OUT]
  const float* be  = (const float*)d_in[2];   // [E*D_OUT]
  const float* Wr1 = (const float*)d_in[3];   // [D_IN, 2*D_IN]
  const float* br1 = (const float*)d_in[4];   // [2*D_IN]
  const float* Wr2 = (const float*)d_in[5];   // [2*D_IN, E]
  const float* br2 = (const float*)d_in[6];   // [E]
  float* out = (float*)d_out;                 // [B, D_OUT]

  // Workspace layout (256B-aligned chunks)
  char* ws = (char*)d_ws;
  int*    idx   = (int*)(ws);                              //  8192 ints
  int*    cnt16 = (int*)(ws + 32768);                      //  16 ints
  int*    perm  = (int*)(ws + 33024);                      //  PERM_N ints
  __bf16* Wr1T  = (__bf16*)(ws + 66560);                   //  D_HID*D_IN bf16
  __bf16* WeT   = (__bf16*)(ws + 66560 + (size_t)D_HID * D_IN * 2);

  // 1) fp32 -> bf16 transposed weights
  k_transpose_bf16<<<dim3(D_HID / 32, D_IN / 32), 256, 0, stream>>>(
      Wr1, Wr1T, D_IN, D_HID);
  k_transpose_bf16<<<dim3((N_EXP * D_OUT) / 32, D_IN / 32), 256, 0, stream>>>(
      We, WeT, D_IN, N_EXP * D_OUT);

  // 2) init sort state
  k_init<<<(PERM_N + 255) / 256, 256, 0, stream>>>(cnt16, perm);

  // 3) router -> top-1 expert index per row
  k_router<<<B_ROWS / 16, 256, 0, stream>>>(x, Wr1T, br1, Wr2, br2, idx);

  // 4-6) counting sort into padded expert segments
  k_count  <<<B_ROWS / 256, 256, 0, stream>>>(idx, cnt16);
  k_offsets<<<1, 1, 0, stream>>>(cnt16);
  k_scatter<<<B_ROWS / 256, 256, 0, stream>>>(idx, cnt16, perm);

  // 7) grouped single-expert GEMM (scatter outputs by row)
  k_expert<<<N_TILES, 256, 0, stream>>>(x, WeT, be, idx, perm, out);
}
